// GravNetBlock_3470333575659
// MI455X (gfx1250) — matrix operand order repaired and checked by reading, not compiled, not run on hardware
//
#include <hip/hip_runtime.h>
#include <hip/hip_bf16.h>

// ---------------------------------------------------------------------------
// GravNet block for MI455X (gfx1250): f16 WMMA GEMMs (f32 accumulate),
// wave32 top-K selection in LDS, fused bias/tanh epilogues, BN as fused
// scale/shift. All tensors L2-resident (~92MB workspace < 192MB L2).
// ---------------------------------------------------------------------------

#define NGRAPH 16
#define GSIZE  2048
#define NTOT   32768      // NGRAPH*GSIZE
#define KNN    40
#define PD     22

typedef __attribute__((ext_vector_type(16))) _Float16 v16h;
typedef __attribute__((ext_vector_type(8)))  float    v8f;
typedef __attribute__((ext_vector_type(4)))  float    f32x4;

union HalfVec { v16h v; _Float16 h[16]; };

// ------------------------- small utility kernels ---------------------------

__global__ void fill_zero_h(_Float16* p, long long n) {
  long long i = (long long)blockIdx.x * blockDim.x + threadIdx.x;
  if (i < n) p[i] = (_Float16)0.0f;
}

// Pack W [K, Nout] (f32 row-major, row stride ldn) -> Wt [Nout, Kpad] f16,
// writing rows [rowOff, rowOff+Nout). k >= K zero-padded.
__global__ void pack_w(const float* __restrict__ W, int K, int Nout, int ldn,
                       _Float16* __restrict__ Wt, int rowOff, int Kpad) {
  int i = blockIdx.x * blockDim.x + threadIdx.x;
  if (i >= Nout * Kpad) return;
  int n = i / Kpad, k = i % Kpad;
  float v = (k < K) ? W[(long long)k * ldn + n] : 0.0f;
  Wt[(long long)(rowOff + n) * Kpad + k] = (_Float16)v;
}

__global__ void build_bsh(const float* __restrict__ bs, const float* __restrict__ bh,
                          float* __restrict__ bsh) {
  int i = threadIdx.x;              // 32 threads
  float v = 0.0f;
  if (i < 4) v = bs[i];
  else if (i < 26) v = bh[i - 4];
  bsh[i] = v;
}

// x f32 [N,96] -> A0 f16 [N,160]; cols 0..95 = x, 140..159 = 0 (96..139 by agg)
__global__ void encode_x(const float* __restrict__ x, _Float16* __restrict__ A0) {
  long long i = (long long)blockIdx.x * blockDim.x + threadIdx.x;
  if (i >= (long long)NTOT * 160) return;
  int n = (int)(i / 160), c = (int)(i % 160);
  if (c < 96)        A0[i] = (_Float16)x[(long long)n * 96 + c];
  else if (c >= 140) A0[i] = (_Float16)0.0f;
}

// ------------------------- generic WMMA GEMM -------------------------------
// Y[N,NOUT] = act(A[N,KDIM (stride lda)] * Bt[NOUT,KDIM]^T + bias)
// One wave per 16-row strip; NOUT/16 f32 accumulators.

template <int KDIM, int NOUT, bool TANH>
__global__ __launch_bounds__(256)
void gemm_wmma(const _Float16* __restrict__ A, int lda,
               const _Float16* __restrict__ Bt,
               const float* __restrict__ bias,
               float* __restrict__ out, int nrows) {
  constexpr int NT = NOUT / 16;
  int wid  = (blockIdx.x * blockDim.x + threadIdx.x) >> 5;  // global wave id
  int lane = threadIdx.x & 31;
  int m0 = wid * 16;
  if (m0 >= nrows) return;                 // wave-uniform: EXEC stays all-ones
  int hi = lane >> 4;                      // half of the wave (K interleave)
  int mr = lane & 15;                      // row (A) / column (B,C) index

  const _Float16* arow = A + (long long)(m0 + mr) * lda;

  v8f acc[NT];
  v8f vz = {0.f, 0.f, 0.f, 0.f, 0.f, 0.f, 0.f, 0.f};
#pragma unroll
  for (int t = 0; t < NT; t++) acc[t] = vz;

#pragma unroll
  for (int k = 0; k < KDIM; k += 32) {
    if (k + 32 < KDIM) __builtin_prefetch(arow + k + 32, 0, 1);
    // A fragment: 16-bit A 16x32 layout. lane<16: K = [k..k+7],[k+16..k+23];
    // lane>=16: K = [k+8..k+15],[k+24..k+31]. Two contiguous 16B chunks.
    HalfVec ua;
    *(f32x4*)&ua.h[0] = *(const f32x4*)(arow + k + hi * 8);
    *(f32x4*)&ua.h[8] = *(const f32x4*)(arow + k + hi * 8 + 16);
#pragma unroll
    for (int t = 0; t < NT; t++) {
      // B fragment: same K interleave as A. lane<16: K=[k..k+7],[k+16..k+23];
      // lane>=16: K=[k+8..k+15],[k+24..k+31]. Bt is pre-transposed.
      HalfVec ub;
      const _Float16* brow = Bt + (long long)(t * 16 + mr) * KDIM + k + hi * 8;
      *(f32x4*)&ub.h[0] = *(const f32x4*)(brow);
      *(f32x4*)&ub.h[8] = *(const f32x4*)(brow + 16);
      acc[t] = __builtin_amdgcn_wmma_f32_16x16x32_f16(
          /*neg_a=*/false, ua.v, /*neg_b=*/false, ub.v,
          /*c_mod=*/(short)0, acc[t], /*reuse_a=*/false, /*reuse_b=*/false);
    }
  }

  // C/D layout: lane<16 -> M=j, N=lane; lane>=16 -> M=8+j, N=lane-16.
#pragma unroll
  for (int t = 0; t < NT; t++) {
#pragma unroll
    for (int j = 0; j < 8; j++) {
      int m = hi * 8 + j;
      int n = t * 16 + mr;
      float val = acc[t][j] + bias[n];
      if (TANH) val = tanhf(val);
      out[(long long)(m0 + m) * NOUT + n] = val;
    }
  }
}

// ------------------------- kNN top-40 selection ----------------------------
// One wave per hit; wave-private 8KB LDS slice holds d^2 to all 2048
// in-graph candidates; 40 rounds of lane-argmin + wave32 shuffle reduction.

__global__ __launch_bounds__(128)
void topk_kernel(const float* __restrict__ sh,   // [N,32], s in cols 0..3
                 int* __restrict__ idx, float* __restrict__ wgt) {
  __shared__ float dbuf[4][GSIZE];
  int wave = threadIdx.x >> 5, lane = threadIdx.x & 31;
  int r = blockIdx.x * 4 + wave;
  int g = r >> 11, base = g << 11;
  float* d = dbuf[wave];

  float s0 = sh[(long long)r * 32 + 0];
  float s1 = sh[(long long)r * 32 + 1];
  float s2 = sh[(long long)r * 32 + 2];
  float s3 = sh[(long long)r * 32 + 3];

  for (int j = lane; j < GSIZE; j += 32) {
    const float* sj = sh + (long long)(base + j) * 32;
    float a = sj[0] - s0, b = sj[1] - s1, c = sj[2] - s2, e = sj[3] - s3;
    d[j] = a * a + b * b + c * c + e * e;   // >= 0 by construction
  }
  // wave-private LDS region: no cross-wave barrier needed

  for (int k = 0; k < KNN; k++) {
    float best = 3.4e38f; int bi = 0;
    for (int j = lane; j < GSIZE; j += 32) {
      float v = d[j];
      if (v < best) { best = v; bi = j; }
    }
#pragma unroll
    for (int off = 16; off > 0; off >>= 1) {
      float ob = __shfl_xor(best, off, 32);
      int   oi = __shfl_xor(bi,   off, 32);
      if (ob < best || (ob == best && oi < bi)) { best = ob; bi = oi; }
    }
    // the owning lane (the one that rescans slot bi) knocks it out
    if ((bi & 31) == lane) d[bi] = 3.4e38f;
    if (lane == 0) {
      idx[(long long)r * KNN + k] = base + bi;
      wgt[(long long)r * KNN + k] = expf(-10.0f * best);
    }
  }
}

// ------------------------- neighbor aggregation ----------------------------
// One wave per hit; lanes 0..21 own PD features; mean+max over K=40 messages.

__global__ __launch_bounds__(256)
void agg_kernel(const float* __restrict__ sh,   // h in cols 4..25
                const int* __restrict__ idx, const float* __restrict__ wgt,
                _Float16* __restrict__ A0) {
  int wid  = (blockIdx.x * blockDim.x + threadIdx.x) >> 5;
  int lane = threadIdx.x & 31;
  if (wid >= NTOT) return;
  float sum = 0.0f, mx = -3.4e38f;
  for (int k = 0; k < KNN; k++) {
    int   j = idx[(long long)wid * KNN + k];
    float w = wgt[(long long)wid * KNN + k];
    float hv = (lane < PD) ? sh[(long long)j * 32 + 4 + lane] : 0.0f;
    float m = hv * w;
    sum += m;
    mx = fmaxf(mx, m);
  }
  if (lane < PD) {
    A0[(long long)wid * 160 + 96 + lane]  = (_Float16)(sum * (1.0f / KNN));
    A0[(long long)wid * 160 + 118 + lane] = (_Float16)mx;
  }
}

// ------------------------- BatchNorm stats + apply -------------------------

__global__ __launch_bounds__(256)
void stats_kernel(const float* __restrict__ v, int C,
                  const float* __restrict__ g, const float* __restrict__ be,
                  float* __restrict__ ss) {   // ss[c]=scale, ss[C+c]=shift
  int c = blockIdx.x;
  __shared__ float s1[256], s2[256];
  float a = 0.0f, b = 0.0f;
  for (int n = threadIdx.x; n < NTOT; n += 256) {
    float x = v[(long long)n * C + c];
    a += x; b += x * x;
  }
  s1[threadIdx.x] = a; s2[threadIdx.x] = b;
  __syncthreads();
  for (int o = 128; o > 0; o >>= 1) {
    if (threadIdx.x < o) {
      s1[threadIdx.x] += s1[threadIdx.x + o];
      s2[threadIdx.x] += s2[threadIdx.x + o];
    }
    __syncthreads();
  }
  if (threadIdx.x == 0) {
    float m = s1[0] * (1.0f / NTOT);
    float var = s2[0] * (1.0f / NTOT) - m * m;
    float rs = rsqrtf(var + 1e-5f);
    float sc = g[c] * rs;
    ss[c] = sc;
    ss[C + c] = be[c] - m * sc;
  }
}

__global__ void bn_f16(const float* __restrict__ v, const float* __restrict__ ss,
                       int C, _Float16* __restrict__ out) {
  long long i = (long long)blockIdx.x * blockDim.x + threadIdx.x;
  if (i >= (long long)NTOT * C) return;
  int c = (int)(i % C);
  out[i] = (_Float16)(v[i] * ss[c] + ss[C + c]);
}

__global__ void bn_f32(const float* __restrict__ v, const float* __restrict__ ss,
                       int C, float* __restrict__ out) {
  long long i = (long long)blockIdx.x * blockDim.x + threadIdx.x;
  if (i >= (long long)NTOT * C) return;
  int c = (int)(i % C);
  out[i] = v[i] * ss[c] + ss[C + c];
}

// ------------------------- global exchange ---------------------------------

__global__ __launch_bounds__(256)
void gex_kernel(const float* __restrict__ v, _Float16* __restrict__ mmm) {
  int g = blockIdx.x / 96, c = blockIdx.x % 96;
  __shared__ float sa[256], sn[256], sx[256];
  float a = 0.0f, mn = 3.4e38f, mx = -3.4e38f;
  for (int j = threadIdx.x; j < GSIZE; j += 256) {
    float x = v[((long long)g * GSIZE + j) * 96 + c];
    a += x; mn = fminf(mn, x); mx = fmaxf(mx, x);
  }
  sa[threadIdx.x] = a; sn[threadIdx.x] = mn; sx[threadIdx.x] = mx;
  __syncthreads();
  for (int o = 128; o > 0; o >>= 1) {
    if (threadIdx.x < o) {
      sa[threadIdx.x] += sa[threadIdx.x + o];
      sn[threadIdx.x] = fminf(sn[threadIdx.x], sn[threadIdx.x + o]);
      sx[threadIdx.x] = fmaxf(sx[threadIdx.x], sx[threadIdx.x + o]);
    }
    __syncthreads();
  }
  if (threadIdx.x == 0) {
    mmm[(long long)g * 288 + c]       = (_Float16)(sa[0] * (1.0f / GSIZE));
    mmm[(long long)g * 288 + 96 + c]  = (_Float16)sn[0];
    mmm[(long long)g * 288 + 192 + c] = (_Float16)sx[0];
  }
}

__global__ void feat_kernel(const float* __restrict__ v, const _Float16* __restrict__ mmm,
                            _Float16* __restrict__ H2) {
  long long i = (long long)blockIdx.x * blockDim.x + threadIdx.x;
  if (i >= (long long)NTOT * 384) return;
  int n = (int)(i / 384), c = (int)(i % 384);
  int g = n >> 11;
  H2[i] = (c < 288) ? mmm[(long long)g * 288 + c]
                    : (_Float16)v[(long long)n * 96 + (c - 288)];
}

// ------------------------- host orchestration ------------------------------

extern "C" void kernel_launch(void* const* d_in, const int* in_sizes, int n_in,
                              void* d_out, int out_size, void* d_ws, size_t ws_size,
                              hipStream_t stream) {
  const float* x   = (const float*)d_in[0];
  const float* Ws  = (const float*)d_in[2];
  const float* bs  = (const float*)d_in[3];
  const float* Wh  = (const float*)d_in[4];
  const float* bh  = (const float*)d_in[5];
  const float* Wo  = (const float*)d_in[6];
  const float* bo  = (const float*)d_in[7];
  const float* g1  = (const float*)d_in[8];
  const float* be1 = (const float*)d_in[9];
  const float* W1  = (const float*)d_in[10];
  const float* b1  = (const float*)d_in[11];
  const float* g2  = (const float*)d_in[12];
  const float* be2 = (const float*)d_in[13];
  const float* W2  = (const float*)d_in[14];
  const float* b2  = (const float*)d_in[15];
  const float* W3  = (const float*)d_in[16];
  const float* b3  = (const float*)d_in[17];
  const float* g3  = (const float*)d_in[18];
  const float* be3 = (const float*)d_in[19];

  const long long N = NTOT;
  char* ws = (char*)d_ws;
  size_t off = 0;
  auto alloc = [&](size_t bytes) -> size_t {
    size_t o = off;
    off = (off + bytes + 255) & ~(size_t)255;
    return o;
  };

  size_t o_wsh = alloc((size_t)32  * 96  * 2);
  size_t o_wo  = alloc((size_t)96  * 160 * 2);
  size_t o_w1  = alloc((size_t)128 * 96  * 2);
  size_t o_w2  = alloc((size_t)96  * 128 * 2);
  size_t o_w3  = alloc((size_t)96  * 384 * 2);
  size_t wzero = off;                       // zero [0, wzero) once per call
  size_t o_bsh = alloc(32 * 4);
  size_t o_ss1 = alloc(256 * 4);
  size_t o_ss2 = alloc(256 * 4);
  size_t o_ss3 = alloc(256 * 4);
  size_t o_mmm = alloc((size_t)NGRAPH * 288 * 2);
  size_t o_sh  = alloc((size_t)N * 32  * 4);
  size_t o_A0  = alloc((size_t)N * 160 * 2);
  size_t o_idx = alloc((size_t)N * KNN * 4);
  size_t o_w   = alloc((size_t)N * KNN * 4);
  size_t o_F1  = alloc((size_t)N * 128 * 4);
  size_t o_F2  = alloc((size_t)N * 128 * 4);
  size_t o_H1  = alloc((size_t)N * 128 * 2);
  size_t o_H2  = alloc((size_t)N * 384 * 2);
  (void)ws_size; (void)in_sizes; (void)n_in; (void)out_size;

  _Float16* WshT = (_Float16*)(ws + o_wsh);
  _Float16* WoT  = (_Float16*)(ws + o_wo);
  _Float16* W1T  = (_Float16*)(ws + o_w1);
  _Float16* W2T  = (_Float16*)(ws + o_w2);
  _Float16* W3T  = (_Float16*)(ws + o_w3);
  float*    bsh  = (float*)(ws + o_bsh);
  float*    ss1  = (float*)(ws + o_ss1);
  float*    ss2  = (float*)(ws + o_ss2);
  float*    ss3  = (float*)(ws + o_ss3);
  _Float16* mmm  = (_Float16*)(ws + o_mmm);
  float*    SH   = (float*)(ws + o_sh);
  _Float16* A0   = (_Float16*)(ws + o_A0);
  int*      IDX  = (int*)(ws + o_idx);
  float*    WGT  = (float*)(ws + o_w);
  float*    F1   = (float*)(ws + o_F1);
  float*    F2   = (float*)(ws + o_F2);
  _Float16* H1   = (_Float16*)(ws + o_H1);
  _Float16* H2   = (_Float16*)(ws + o_H2);

  auto cdiv = [](long long a, long long b) { return (int)((a + b - 1) / b); };

  // --- weight packing (f16, transposed, K padded) ---
  long long nzh = (long long)(wzero / 2);
  fill_zero_h<<<cdiv(nzh, 256), 256, 0, stream>>>((_Float16*)ws, nzh);
  pack_w<<<cdiv(4   * 96,  256), 256, 0, stream>>>(Ws, 96,  4,   4,   WshT, 0, 96);
  pack_w<<<cdiv(22  * 96,  256), 256, 0, stream>>>(Wh, 96,  22,  22,  WshT, 4, 96);
  pack_w<<<cdiv(96  * 160, 256), 256, 0, stream>>>(Wo, 140, 96,  96,  WoT,  0, 160);
  pack_w<<<cdiv(128 * 96,  256), 256, 0, stream>>>(W1, 96,  128, 128, W1T,  0, 96);
  pack_w<<<cdiv(96  * 128, 256), 256, 0, stream>>>(W2, 128, 96,  96,  W2T,  0, 128);
  pack_w<<<cdiv(96  * 384, 256), 256, 0, stream>>>(W3, 384, 96,  96,  W3T,  0, 384);
  build_bsh<<<1, 32, 0, stream>>>(bs, bh, bsh);

  // --- stage x into the f16 concat buffer ---
  encode_x<<<cdiv(N * 160, 256), 256, 0, stream>>>(x, A0);

  const int gemm_blocks = (int)(N / 16 / 8);  // 8 waves per 256-thread block

  // --- GravNetConv: s,h = x @ [Ws|Wh] ---
  gemm_wmma<96, 32, false><<<gemm_blocks, 256, 0, stream>>>(A0, 160, WshT, bsh, SH, (int)N);
  // --- kNN + gaussian weights ---
  topk_kernel<<<(int)(N / 4), 128, 0, stream>>>(SH, IDX, WGT);
  // --- weighted mean/max aggregation -> A0 cols 96..139 ---
  agg_kernel<<<cdiv(N * 32, 256), 256, 0, stream>>>(SH, IDX, WGT, A0);
  // --- output lin of GravNetConv ---
  gemm_wmma<160, 96, false><<<gemm_blocks, 256, 0, stream>>>(A0, 160, WoT, bo, F1, (int)N);

  // --- BN1 -> tanh(W1) ---
  stats_kernel<<<96, 256, 0, stream>>>(F1, 96, g1, be1, ss1);
  bn_f16<<<cdiv(N * 96, 256), 256, 0, stream>>>(F1, ss1, 96, H1);
  gemm_wmma<96, 128, true><<<gemm_blocks, 256, 0, stream>>>(H1, 96, W1T, b1, F2, (int)N);

  // --- BN2 -> tanh(W2) ---
  stats_kernel<<<128, 256, 0, stream>>>(F2, 128, g2, be2, ss2);
  bn_f16<<<cdiv(N * 128, 256), 256, 0, stream>>>(F2, ss2, 128, H1);
  gemm_wmma<128, 96, true><<<gemm_blocks, 256, 0, stream>>>(H1, 128, W2T, b2, F1, (int)N);

  // --- global exchange + concat ---
  gex_kernel<<<NGRAPH * 96, 256, 0, stream>>>(F1, mmm);
  feat_kernel<<<cdiv(N * 384, 256), 256, 0, stream>>>(F1, mmm, H2);

  // --- output head: tanh(W3) -> BN3 -> out ---
  gemm_wmma<384, 96, true><<<gemm_blocks, 256, 0, stream>>>(H2, 384, W3T, b3, F2, (int)N);
  stats_kernel<<<96, 256, 0, stream>>>(F2, 96, g3, be3, ss3);
  bn_f32<<<cdiv(N * 96, 256), 256, 0, stream>>>(F2, ss3, 96, (float*)d_out);
}
